// CausalWanSelfAttention_77025943486666
// MI455X (gfx1250) — compile-verified, matrix operand order checked by reading, not run on hardware
//
#include <hip/hip_runtime.h>
#include <hip/hip_bf16.h>
#include <cstddef>
#include <cstdint>

// ---------------- constants (module-level constants of the reference) -------
#define S_TOK   1560      // F*H*W tokens
#define DIMC    1536
#define NHC     12
#define HDC     128
#define MAX_ATTN_C 32760
#define LCAP    9376      // padded capacity of the attention window (>= 9360)

typedef __attribute__((ext_vector_type(16))) _Float16 v16h;
typedef __attribute__((ext_vector_type(8)))  _Float16 v8h;
typedef __attribute__((ext_vector_type(8)))  float    v8f;
typedef unsigned int u32x4 __attribute__((ext_vector_type(4)));
typedef int          i32x4 __attribute__((ext_vector_type(4)));
typedef int          i32x8 __attribute__((ext_vector_type(8)));

#if defined(__has_builtin)
#if __has_builtin(__builtin_amdgcn_tensor_load_to_lds) && \
    __has_builtin(__builtin_amdgcn_s_wait_tensorcnt)
#define USE_TDM 1
#endif
#endif
#ifndef USE_TDM
#define USE_TDM 0
#endif

#if USE_TDM
#pragma message("CDNA5 probe: TDM tensor_load_to_lds path ENABLED")
#else
#pragma message("CDNA5 probe: TDM builtin unavailable, vector-copy staging")
#endif

// ---------------------------------------------------------------------------
// Fragment load, CDNA5 wave32 WMMA f16 layout (CDNA5 ISA §7.12.2).
// A 16x32 (MxK): lane L -> M = L&15, half = L>>4;
//   VGPR0..3 : K = half*8 + 0..7,  VGPR4..7 : K = 16 + half*8 + 0..7.
// B fragments use the mirror layout (load row-major B^T identically).
// ---------------------------------------------------------------------------
__device__ __forceinline__ v16h ldfrag(const _Float16* p, int ld, int lane) {
  const int m    = lane & 15;
  const int half = lane >> 4;
  const v8h lo = *(const v8h*)(p + m * ld + half * 8);
  const v8h hi = *(const v8h*)(p + m * ld + 16 + half * 8);
  v16h a;
#pragma unroll
  for (int i = 0; i < 8; ++i) { a[i] = lo[i]; a[i + 8] = hi[i]; }
  return a;
}

__device__ __forceinline__ v8f wmma_f16(v16h a, v16h b, v8f c) {
  return __builtin_amdgcn_wmma_f32_16x16x32_f16(false, a, false, b,
                                                (short)0, c, false, false);
}

// Window geometry, computed device-side from current_start.
struct Win { int wstart, Lwin, ncache, Lpad; };
__device__ __forceinline__ Win win_params(int cur) {
  Win w;
  const int lend = cur + S_TOK;
  w.wstart = lend - MAX_ATTN_C; if (w.wstart < 0) w.wstart = 0;
  w.Lwin   = lend - w.wstart;
  w.ncache = cur - w.wstart;
  int lp = (w.Lwin + 31) & ~31;
  w.Lpad  = lp > LCAP ? LCAP : lp;
  if (w.Lwin > w.Lpad) w.Lwin = w.Lpad;   // capacity clamp (not hit for CUR=7800)
  return w;
}

#if USE_TDM
// ---------------------------------------------------------------------------
// Tensor Data Mover: 2D tile (f16 elements) global -> LDS.
// D# bitfields per CDNA5 ISA.
// ---------------------------------------------------------------------------
__device__ __forceinline__ void tdm_load_2d(unsigned lds_off, const void* gptr,
                                            unsigned tensor_w, unsigned tensor_h,
                                            unsigned tile_w, unsigned tile_h,
                                            unsigned row_stride_elems) {
  const unsigned long long ga = (unsigned long long)(uintptr_t)gptr;
  u32x4 g0;
  g0[0] = 1u;                                   // count=1, user descriptor
  g0[1] = lds_off;                              // lds_addr (bytes)
  g0[2] = (unsigned)(ga & 0xffffffffu);         // global_addr[31:0]
  g0[3] = (unsigned)((ga >> 32) & 0x01ffffffu)  // global_addr[56:32]
        | (2u << 30);                           // type = 2 ("image")
  i32x8 g1;
  g1[0] = (int)(1u << 16);                      // workgroup_mask=0, data_size=1 (2B)
  g1[1] = (int)((tensor_w & 0xffffu) << 16);    // tensor_dim0[15:0] @ bits 63:48
  g1[2] = (int)(((tensor_w >> 16) & 0xffffu) |  // tensor_dim0[31:16]
                ((tensor_h & 0xffffu) << 16));  // tensor_dim1[15:0]
  g1[3] = (int)(((tensor_h >> 16) & 0xffffu) |  // tensor_dim1[31:16]
                ((tile_w & 0xffffu) << 16));    // tile_dim0
  g1[4] = (int)(tile_h & 0xffffu);              // tile_dim1 (tile_dim2 = 0)
  g1[5] = (int)row_stride_elems;                // tensor_dim0_stride[31:0]
  g1[6] = 0;                                    // stride hi / dim1_stride lo
  g1[7] = 0;
  const i32x4 z4 = {0, 0, 0, 0};
#if __clang_major__ >= 23
  const i32x8 z8 = {0, 0, 0, 0, 0, 0, 0, 0};
  __builtin_amdgcn_tensor_load_to_lds(g0, g1, z4, z4, z8, 0);
#else
  __builtin_amdgcn_tensor_load_to_lds(g0, g1, z4, z4, 0);
#endif
}
#endif  // USE_TDM

// ---------------------------------------------------------------------------
// Tiled GEMM: out[M,N] = A[M,K] * B[K,N] + bias[N]  (f16 WMMA, f32 accumulate)
// 256 threads = 8 waves; 128x64 block tile; each wave computes 32x32
// (2 A-frags x 2 B-frags -> 4 WMMAs per 32-wide K step).
// ---------------------------------------------------------------------------
template <typename AT>
__global__ __launch_bounds__(256)
void gemm16(const AT* __restrict__ A, const float* __restrict__ Bm,
            const float* __restrict__ bias, float* __restrict__ out,
            int M, int N, int K) {
  __shared__ __align__(16) _Float16 As [128][32];
  __shared__ __align__(16) _Float16 Bts[64][32];   // B transposed: [n][k]

  const int tid  = threadIdx.x;
  const int lane = tid & 31;
  const int wave = tid >> 5;
  const int mw   = wave >> 1;   // 0..3 : 32-row slab
  const int nw   = wave & 1;    // 0..1 : 32-col slab
  const int bm   = blockIdx.y * 128;
  const int bn   = blockIdx.x * 64;

  v8f c00 = {}, c01 = {}, c10 = {}, c11 = {};

  for (int k0 = 0; k0 < K; k0 += 32) {
    // --- stage A tile (128 x 32), convert to f16 ---
    {
      const int row = tid >> 1;          // 0..127
      const int kk  = (tid & 1) * 16;    // 0 or 16
      const int gm  = bm + row;
      _Float16* dst = &As[row][kk];
      if (gm < M) {
        const AT* src = A + (size_t)gm * K + k0 + kk;
        __builtin_prefetch(src + 32, 0, 0);
#pragma unroll
        for (int j = 0; j < 16; ++j) dst[j] = (_Float16)src[j];
      } else {
#pragma unroll
        for (int j = 0; j < 16; ++j) dst[j] = (_Float16)0.0f;
      }
    }
    // --- stage B tile transposed (32k x 64n -> Bts[n][k]) ---
    {
      const int krow = tid >> 3;         // 0..31
      const int nc0  = (tid & 7) * 8;    // 0..56
      const float* src = Bm + (size_t)(k0 + krow) * N + bn + nc0;
#pragma unroll
      for (int j = 0; j < 8; ++j) Bts[nc0 + j][krow] = (_Float16)src[j];
    }
    __syncthreads();

    const v16h a0 = ldfrag(&As [mw * 32     ][0], 32, lane);
    const v16h a1 = ldfrag(&As [mw * 32 + 16][0], 32, lane);
    const v16h b0 = ldfrag(&Bts[nw * 32     ][0], 32, lane);
    const v16h b1 = ldfrag(&Bts[nw * 32 + 16][0], 32, lane);
    c00 = wmma_f16(a0, b0, c00);
    c01 = wmma_f16(a0, b1, c01);
    c10 = wmma_f16(a1, b0, c10);
    c11 = wmma_f16(a1, b1, c11);
    __syncthreads();
  }

  const int half = lane >> 4;
  const int n    = lane & 15;
#pragma unroll
  for (int r = 0; r < 8; ++r) {
    const int gm0 = bm + mw * 32 + half * 8 + r;
    const int gm1 = gm0 + 16;
    const int gn0 = bn + nw * 32 + n;
    const int gn1 = gn0 + 16;
    if (gm0 < M) {
      out[(size_t)gm0 * N + gn0] = c00[r] + bias[gn0];
      out[(size_t)gm0 * N + gn1] = c01[r] + bias[gn1];
    }
    if (gm1 < M) {
      out[(size_t)gm1 * N + gn0] = c10[r] + bias[gn0];
      out[(size_t)gm1 * N + gn1] = c11[r] + bias[gn1];
    }
  }
}

// ---------------------------------------------------------------------------
// Window build: convert the cached part of the K/V attention window to f16.
//   kw [NH][Lpad][HD]   (K rows per head, row-major)
//   vwt[NH*HD][Lpad]    (V transposed per head)
// New tokens (keys ncache..Lwin) are written by rms_rope_kernel; tail keys
// (Lwin..Lpad) are zero-filled here.
// ---------------------------------------------------------------------------
__global__ __launch_bounds__(256)
void build_window(const float* __restrict__ ck, const float* __restrict__ cv,
                  const int* __restrict__ curp,
                  _Float16* __restrict__ kw, _Float16* __restrict__ vwt) {
  const Win W = win_params(curp[0]);
  const long long total = (long long)W.Lpad * NHC * HDC;
  const long long step  = (long long)gridDim.x * blockDim.x;
  for (long long idx = (long long)blockIdx.x * blockDim.x + threadIdx.x;
       idx < total; idx += step) {
    const int d = (int)(idx & (HDC - 1));
    const int h = (int)((idx >> 7) % NHC);
    const int k = (int)(idx / (NHC * HDC));
    if (k >= W.ncache && k < W.Lwin) continue;   // new tokens: rms_rope writes
    float kvv = 0.f, vvv = 0.f;
    if (k < W.ncache) {
      const size_t co = ((size_t)(W.wstart + k) * NHC + h) * HDC + d;
      kvv = ck[co];
      vvv = cv[co];
    }
    kw [((size_t)h * W.Lpad + k) * HDC + d] = (_Float16)kvv;
    vwt[((size_t)h * HDC + d) * W.Lpad + k] = (_Float16)vvv;
  }
}

// ---------------------------------------------------------------------------
// RMSNorm (full 1536-dim row) + gain + 3D RoPE.  One block per token.
// Q -> qf (f16, pre-scaled by 1/sqrt(HD)); K -> kw slot; V -> vwt slot.
// ---------------------------------------------------------------------------
__global__ __launch_bounds__(256)
void rms_rope_kernel(const float* __restrict__ qraw, const float* __restrict__ kraw,
                     const float* __restrict__ vraw,
                     const float* __restrict__ gq, const float* __restrict__ gk,
                     const float* __restrict__ fcos, const float* __restrict__ fsin,
                     const int* __restrict__ hp, const int* __restrict__ wp,
                     const int* __restrict__ curp,
                     _Float16* __restrict__ qf, _Float16* __restrict__ kw,
                     _Float16* __restrict__ vwt) {
  const int t   = blockIdx.x;
  const int tid = threadIdx.x;
  const float* qr = qraw + (size_t)t * DIMC;
  const float* kr = kraw + (size_t)t * DIMC;

  float sq = 0.f, sk = 0.f;
  for (int d = tid; d < DIMC; d += 256) {
    const float a = qr[d]; sq += a * a;
    const float b = kr[d]; sk += b * b;
  }
#pragma unroll
  for (int off = 16; off; off >>= 1) {
    sq += __shfl_xor(sq, off, 32);
    sk += __shfl_xor(sk, off, 32);
  }
  __shared__ float sqs[8], sks[8];
  if ((tid & 31) == 0) { sqs[tid >> 5] = sq; sks[tid >> 5] = sk; }
  __syncthreads();
  float tq = 0.f, tk = 0.f;
#pragma unroll
  for (int i = 0; i < 8; ++i) { tq += sqs[i]; tk += sks[i]; }
  const float rq = rsqrtf(tq / (float)DIMC + 1e-6f);
  const float rk = rsqrtf(tk / (float)DIMC + 1e-6f);

  const int Hh    = hp[0];
  const int Ww    = wp[0];
  const int frame = Hh * Ww;
  const int cur   = curp[0];
  const Win W     = win_params(cur);
  const int key   = W.ncache + t;               // this token's window slot
  const int sf    = cur / frame;                // start_frame
  const int fidx  = t / frame;
  const int hh    = (t % frame) / Ww;
  const int ww    = t % Ww;
  const float qscale = 0.08838834764831845f;    // 1/sqrt(128)

  // C = 64 angle channels per head; c0 = 22 (frame), then 21 (h), 21 (w)
  for (int p = tid; p < DIMC / 2; p += 256) {
    const int head = p >> 6;
    const int c    = p & 63;
    const int row  = (c < 22) ? (sf + fidx) : ((c < 43) ? hh : ww);
    const float cs = fcos[row * 64 + c];
    const float sn = fsin[row * 64 + c];
    const int d0 = head * HDC + 2 * c;
    const int d1 = d0 + 1;
    const float q0 = qr[d0] * rq * gq[d0], q1 = qr[d1] * rq * gq[d1];
    const float k0 = kr[d0] * rk * gk[d0], k1 = kr[d1] * rk * gk[d1];
    const size_t qo = (size_t)t * DIMC + d0;
    qf[qo]     = (_Float16)((q0 * cs - q1 * sn) * qscale);
    qf[qo + 1] = (_Float16)((q0 * sn + q1 * cs) * qscale);
    const size_t ko = ((size_t)head * W.Lpad + key) * HDC + 2 * c;
    kw[ko]     = (_Float16)(k0 * cs - k1 * sn);
    kw[ko + 1] = (_Float16)(k0 * sn + k1 * cs);
  }
  const float* vr = vraw + (size_t)t * DIMC;
  for (int d = tid; d < DIMC; d += 256)          // vwt flat: [NH*HD][Lpad]
    vwt[(size_t)d * W.Lpad + key] = (_Float16)vr[d];
}

// ---------------------------------------------------------------------------
// Flash attention: 128 threads = 4 waves; each wave owns a 16-query tile
// (block covers 64 queries of one head).  32-key chunks are staged into
// double-buffered LDS by the Tensor Data Mover: chunk ch+1's DMA is issued
// before chunk ch's compute, and the wave waits only for TENSORcnt<=2
// (TDM completes in order), hiding the DMA latency under 16 WMMAs + softmax.
// ---------------------------------------------------------------------------
__global__ __launch_bounds__(128)
void attn_kernel(const _Float16* __restrict__ qf,
                 const _Float16* __restrict__ kw, const _Float16* __restrict__ vwt,
                 const int* __restrict__ curp, _Float16* __restrict__ of) {
  // buffer b at [b*8192, b*8192+8192):  Ks 32x128 | Vts 128x32
  // [16384, 18432): Pt  4 waves x 16 x 32
  __shared__ __align__(16) _Float16 smem[2 * (32 * HDC + HDC * 32) + 4 * 16 * 32];

  const int tid  = threadIdx.x;
  const int lane = tid & 31;
  const int wave = tid >> 5;
  const int h    = blockIdx.y;
  const int m0   = blockIdx.x * 64 + wave * 16;

  const Win W = win_params(curp[0]);
  const _Float16* kwh  = kw  + (size_t)h * W.Lpad * HDC;
  const _Float16* vwth = vwt + (size_t)h * HDC * W.Lpad;
  const int nchunks = W.Lpad >> 5;

  // ---- stage 64 Q rows into buffer 0 area, build per-wave A-fragments ----
  {
    const int row = tid >> 1;           // 0..63
    const int d0  = (tid & 1) * 64;
    const int gm  = blockIdx.x * 64 + row;
    _Float16* dst = smem + row * HDC + d0;
    if (gm < S_TOK) {
      const _Float16* src = qf + (size_t)gm * DIMC + h * HDC + d0;
#pragma unroll
      for (int j = 0; j < 8; ++j) *(v8h*)(dst + j * 8) = *(const v8h*)(src + j * 8);
    } else {
#pragma unroll
      for (int j = 0; j < 64; ++j) dst[j] = (_Float16)0.0f;
    }
  }
  __syncthreads();
  v16h qfr[4];
#pragma unroll
  for (int kc = 0; kc < 4; ++kc)
    qfr[kc] = ldfrag(smem + (wave * 16) * HDC + kc * 32, HDC, lane);
  __syncthreads();   // all Q fragments read before buffer 0 is overwritten

#if USE_TDM
  const unsigned ldsbase = (unsigned)(uintptr_t)(const void*)smem;
  // issue chunk 0 into buffer 0 (2 tensor ops: K tile + Vt tile)
  if (tid < 32) {
    tdm_load_2d(ldsbase, kwh, 32 * HDC, 1, 32 * HDC, 1, 32 * HDC);
    tdm_load_2d(ldsbase + 8192, vwth, 32, HDC, 32, HDC, (unsigned)W.Lpad);
  }
#endif

  v8f ofrag[8];
#pragma unroll
  for (int nt = 0; nt < 8; ++nt) ofrag[nt] = (v8f){};
  float mrow[8], lrow[8];
#pragma unroll
  for (int r = 0; r < 8; ++r) { mrow[r] = -1e30f; lrow[r] = 0.f; }

  const int halfq = lane >> 4;
  const int ncol  = lane & 15;
  _Float16* Ptw   = smem + 16384 + wave * 512;   // 16 x 32 per wave

  for (int ch = 0; ch < nchunks; ++ch) {
    const int kbase = ch << 5;
    const _Float16* Kbuf = smem + (ch & 1) * 8192;
    const _Float16* Vbuf = Kbuf + 4096;

#if USE_TDM
    if (tid < 32) {
      if (ch + 1 < nchunks) {
        // prefetch next chunk into the other buffer, then wait for current
        const unsigned nb = ((unsigned)(ch + 1) & 1u) * 16384u;
        const int nk = (ch + 1) << 5;
        tdm_load_2d(ldsbase + nb, kwh + (size_t)nk * HDC,
                    32 * HDC, 1, 32 * HDC, 1, 32 * HDC);
        tdm_load_2d(ldsbase + nb + 8192, vwth + nk,
                    32, HDC, 32, HDC, (unsigned)W.Lpad);
        __builtin_amdgcn_s_wait_tensorcnt(2);   // current buffer's 2 ops done
      } else {
        __builtin_amdgcn_s_wait_tensorcnt(0);
      }
    }
#else
    {
      _Float16* kdst = smem + (ch & 1) * 8192;
      const _Float16* ksrc = kwh + (size_t)kbase * HDC;  // 4096 f16 contiguous
#pragma unroll
      for (int j = tid; j < 512; j += 128)
        *(v8h*)(kdst + j * 8) = *(const v8h*)(ksrc + j * 8);
      const _Float16* vrow = vwth + kbase + (size_t)tid * W.Lpad;
      _Float16* vdst = kdst + 4096 + tid * 32;
#pragma unroll
      for (int j = 0; j < 4; ++j)
        *(v8h*)(vdst + j * 8) = *(const v8h*)(vrow + j * 8);
    }
#endif
    __syncthreads();

    // ---- S = Q * K^T : preload fragment groups, then WMMA back-to-back ----
    v8f sc0 = {}, sc1 = {};
    {
      v16h bf0[4], bf1[4];
#pragma unroll
      for (int kc = 0; kc < 4; ++kc)
        bf0[kc] = ldfrag(Kbuf + kc * 32, HDC, lane);
#pragma unroll
      for (int kc = 0; kc < 4; ++kc)
        bf1[kc] = ldfrag(Kbuf + 16 * HDC + kc * 32, HDC, lane);
#pragma unroll
      for (int kc = 0; kc < 4; ++kc) {
        sc0 = wmma_f16(qfr[kc], bf0[kc], sc0);
        sc1 = wmma_f16(qfr[kc], bf1[kc], sc1);
      }
    }
    // key-tail mask (uniform branch; active only for the last chunk)
    if (kbase + 32 > W.Lwin) {
      const int k0n = kbase + ncol;
      if (k0n >= W.Lwin) {
#pragma unroll
        for (int r = 0; r < 8; ++r) sc0[r] = -1e30f;
      }
      if (k0n + 16 >= W.Lwin) {
#pragma unroll
        for (int r = 0; r < 8; ++r) sc1[r] = -1e30f;
      }
    }

    // ---- online softmax (each row lives across 16 lanes of a half-wave) ---
    float alpha[8];
#pragma unroll
    for (int r = 0; r < 8; ++r) {
      float v = fmaxf(sc0[r], sc1[r]);
#pragma unroll
      for (int off = 8; off; off >>= 1) v = fmaxf(v, __shfl_xor(v, off, 16));
      const float mnew = fmaxf(mrow[r], v);
      const float al   = __expf(mrow[r] - mnew);
      sc0[r] = __expf(sc0[r] - mnew);
      sc1[r] = __expf(sc1[r] - mnew);
      float sv = sc0[r] + sc1[r];
#pragma unroll
      for (int off = 8; off; off >>= 1) sv += __shfl_xor(sv, off, 16);
      lrow[r] = lrow[r] * al + sv;
      mrow[r] = mnew;
      alpha[r] = al;
    }
#pragma unroll
    for (int nt = 0; nt < 8; ++nt)
#pragma unroll
      for (int r = 0; r < 8; ++r) ofrag[nt][r] *= alpha[r];

    // ---- P tile to LDS, re-fragment as WMMA A operand ----
#pragma unroll
    for (int r = 0; r < 8; ++r) {
      Ptw[(halfq * 8 + r) * 32 + ncol]      = (_Float16)sc0[r];
      Ptw[(halfq * 8 + r) * 32 + 16 + ncol] = (_Float16)sc1[r];
    }
    __syncthreads();

    // ---- O += P * V : preload V fragments in groups of 4 ----
    {
      const v16h pa = ldfrag(Ptw, 32, lane);
#pragma unroll
      for (int g = 0; g < 2; ++g) {
        v16h vf[4];
#pragma unroll
        for (int j = 0; j < 4; ++j)
          vf[j] = ldfrag(Vbuf + ((g * 4 + j) * 16) * 32, 32, lane);
#pragma unroll
        for (int j = 0; j < 4; ++j)
          ofrag[g * 4 + j] = wmma_f16(pa, vf[j], ofrag[g * 4 + j]);
      }
    }
    __syncthreads();   // all waves done with this buffer before re-issue
  }

  // ---- normalize and write O (f16) ----
#pragma unroll
  for (int r = 0; r < 8; ++r) {
    const int gm = m0 + halfq * 8 + r;
    if (gm < S_TOK) {
      const float inv = 1.0f / lrow[r];
#pragma unroll
      for (int nt = 0; nt < 8; ++nt)
        of[(size_t)gm * DIMC + h * HDC + nt * 16 + ncol] =
            (_Float16)(ofrag[nt][r] * inv);
    }
  }
}

// ---------------------------------------------------------------------------
extern "C" void kernel_launch(void* const* d_in, const int* in_sizes, int n_in,
                              void* d_out, int out_size, void* d_ws, size_t ws_size,
                              hipStream_t stream) {
  const float* x  = (const float*)d_in[0];
  const float* wq = (const float*)d_in[1];
  const float* bq = (const float*)d_in[2];
  const float* wk = (const float*)d_in[3];
  const float* bk = (const float*)d_in[4];
  const float* wv = (const float*)d_in[5];
  const float* bv = (const float*)d_in[6];
  const float* wo = (const float*)d_in[7];
  const float* bo = (const float*)d_in[8];
  const float* gq = (const float*)d_in[9];
  const float* gk = (const float*)d_in[10];
  const float* fc = (const float*)d_in[11];
  const float* fs = (const float*)d_in[12];
  const float* cache_k = (const float*)d_in[13];
  const float* cache_v = (const float*)d_in[14];
  const int* h_grid = (const int*)d_in[16];
  const int* w_grid = (const int*)d_in[17];
  const int* cur    = (const int*)d_in[18];
  float* out = (float*)d_out;

  char* ws = (char*)d_ws;
  size_t off = 0;
  auto carve = [&](size_t bytes) -> void* {
    void* p = ws + off;
    off = (off + bytes + 255) & ~(size_t)255;
    return p;
  };
  const size_t n_tok = (size_t)S_TOK * DIMC;
  const size_t n_win = (size_t)NHC * LCAP * HDC;
  float*    qraw = (float*)   carve(n_tok * sizeof(float));
  float*    kraw = (float*)   carve(n_tok * sizeof(float));
  float*    vraw = (float*)   carve(n_tok * sizeof(float));
  _Float16* qf16 = (_Float16*)carve(n_tok * sizeof(_Float16));
  _Float16* of16 = (_Float16*)carve(n_tok * sizeof(_Float16));
  _Float16* kwb  = (_Float16*)carve(n_win * sizeof(_Float16));
  _Float16* vwtb = (_Float16*)carve(n_win * sizeof(_Float16));
  (void)in_sizes; (void)n_in; (void)out_size; (void)ws_size;

  const dim3 gblock(256);
  const dim3 ggrid(DIMC / 64, (S_TOK + 127) / 128);

  // Q/K/V projections (f32 inputs -> f16 WMMA -> f32 accumulate)
  gemm16<float><<<ggrid, gblock, 0, stream>>>(x, wq, bq, qraw, S_TOK, DIMC, DIMC);
  gemm16<float><<<ggrid, gblock, 0, stream>>>(x, wk, bk, kraw, S_TOK, DIMC, DIMC);
  gemm16<float><<<ggrid, gblock, 0, stream>>>(x, wv, bv, vraw, S_TOK, DIMC, DIMC);

  // One-time f16 window build (cache region + zero tail); no input mutation.
  build_window<<<2048, 256, 0, stream>>>(cache_k, cache_v, cur, kwb, vwtb);

  // RMSNorm + RoPE; writes new tokens directly into their window slots.
  rms_rope_kernel<<<S_TOK, 256, 0, stream>>>(qraw, kraw, vraw, gq, gk, fc, fs,
                                             h_grid, w_grid, cur,
                                             qf16, kwb, vwtb);

  // Flash attention: 64 queries x head per block, double-buffered TDM staging.
  attn_kernel<<<dim3((S_TOK + 63) / 64, NHC), 128, 0, stream>>>(
      qf16, kwb, vwtb, cur, of16);

  // Output projection
  gemm16<_Float16><<<ggrid, gblock, 0, stream>>>(of16, wo, bo, out,
                                                 S_TOK, DIMC, DIMC);
}
